// PointNetFeaturePropagation_30056181138030
// MI455X (gfx1250) — matrix operand order repaired and checked by reading, not compiled, not run on hardware
//
#include <hip/hip_runtime.h>

// Problem constants (from setup_inputs): B=8, N=8192, S=2048, D1=64, D2=256
#define B_    8
#define N_    8192
#define S_    2048
#define D1_   64
#define D2_   256
#define CIN_  320
#define C1_   256
#define C2_   128
#define ROWS_ (B_ * N_)   // 65536 rows through both GEMMs

typedef __attribute__((ext_vector_type(16))) __bf16 v16bf;
typedef __attribute__((ext_vector_type(8)))  float  v8f;

__device__ __forceinline__ unsigned short f2bf(float f) {
  unsigned int u = __float_as_uint(f);
  u += 0x7FFFu + ((u >> 16) & 1u);   // round-to-nearest-even
  return (unsigned short)(u >> 16);
}
__device__ __forceinline__ float bf2f(unsigned short s) {
  return __uint_as_float(((unsigned int)s) << 16);
}

// Async global->LDS 16B copy (CDNA5; tracked by ASYNCcnt). LDS offset is the
// low 32 bits of the generic pointer (aperture maps addr[31:0] -> LDS offset).
__device__ __forceinline__ void async_copy_b128(const void* gptr, void* lds_ptr) {
  unsigned lds = (unsigned)(unsigned long long)lds_ptr;
  unsigned long long ga = (unsigned long long)gptr;
  asm volatile("global_load_async_to_lds_b128 %0, %1, off"
               :: "v"(lds), "v"(ga) : "memory");
}
__device__ __forceinline__ void wait_async0() {   // all async copies done
  asm volatile("s_wait_asynccnt 0" ::: "memory");
}
__device__ __forceinline__ void wait_async_le3() { // oldest group (3/thread) done
  asm volatile("s_wait_asynccnt 3" ::: "memory");
}

// ---------------------------------------------------------------------------
// Kernel 1: 3-NN inverse-distance interpolation + concat -> new_points (bf16)
// Grid: (N/256, B), Block: 256. xyz2[b] (24KB) staged in LDS.
// ---------------------------------------------------------------------------
__global__ __launch_bounds__(256) void interp_concat(
    const float* __restrict__ xyz1, const float* __restrict__ xyz2,
    const float* __restrict__ points1, const float* __restrict__ points2,
    unsigned short* __restrict__ NP)   // [ROWS_, CIN_] bf16
{
  __shared__ float sx[S_], sy[S_], sz[S_];
  __shared__ int   sidx[256 * 3];
  __shared__ float swt[256 * 3];
  const int b   = blockIdx.y;
  const int p0  = blockIdx.x * 256;
  const int tid = threadIdx.x;

  for (int s = tid; s < S_; s += 256) {
    const float* q = xyz2 + ((size_t)b * S_ + s) * 3;
    sx[s] = q[0]; sy[s] = q[1]; sz[s] = q[2];
  }
  __syncthreads();

  const int n = p0 + tid;
  const float* p = xyz1 + ((size_t)b * N_ + n) * 3;
  const float px = p[0], py = p[1], pz = p[2];
  float d0 = 3.4e38f, d1 = 3.4e38f, d2 = 3.4e38f;
  int   i0 = 0, i1 = 0, i2 = 0;
  for (int s = 0; s < S_; s++) {
    float dx = px - sx[s], dy = py - sy[s], dz = pz - sz[s];
    float d = fmaf(dx, dx, fmaf(dy, dy, dz * dz));
    if (d < d2) {
      if (d < d1) {
        if (d < d0) { d2 = d1; i2 = i1; d1 = d0; i1 = i0; d0 = d; i0 = s; }
        else        { d2 = d1; i2 = i1; d1 = d;  i1 = s; }
      } else        { d2 = d;  i2 = s; }
    }
  }
  float w0 = 1.f / fmaxf(d0, 1e-10f);
  float w1 = 1.f / fmaxf(d1, 1e-10f);
  float w2 = 1.f / fmaxf(d2, 1e-10f);
  float inv = 1.f / (w0 + w1 + w2);
  sidx[tid * 3 + 0] = i0; sidx[tid * 3 + 1] = i1; sidx[tid * 3 + 2] = i2;
  swt [tid * 3 + 0] = w0 * inv; swt[tid * 3 + 1] = w1 * inv; swt[tid * 3 + 2] = w2 * inv;
  __syncthreads();

  // Coalesced gather: for each point in the block, thread tid owns channel tid.
  const float* P2 = points2 + (size_t)b * S_ * D2_;
  for (int q = 0; q < 256; q++) {
    float v = swt[q * 3 + 0] * P2[(size_t)sidx[q * 3 + 0] * D2_ + tid]
            + swt[q * 3 + 1] * P2[(size_t)sidx[q * 3 + 1] * D2_ + tid]
            + swt[q * 3 + 2] * P2[(size_t)sidx[q * 3 + 2] * D2_ + tid];
    NP[((size_t)b * N_ + p0 + q) * CIN_ + D1_ + tid] = f2bf(v);
  }
  // Copy points1 (first 64 channels) into the concat buffer.
  for (int e = tid; e < 256 * D1_; e += 256) {
    int q = e >> 6, cch = e & 63;
    NP[((size_t)b * N_ + p0 + q) * CIN_ + cch] =
        f2bf(points1[((size_t)b * N_ + p0 + q) * D1_ + cch]);
  }
}

// ---------------------------------------------------------------------------
// GEMM: A[ROWS,K](bf16) x W^T[K,NC](bf16, row-major [NC,K]) -> H + BN partials.
// Block tile 128(M) x 64(N); 8 waves = 4(M) x 2(N), each wave owns a 32x32
// tile (4 WMMAs per k-step). Double-buffered LDS staging via async
// global->LDS b128 copies overlapped with WMMA of the previous tile.
// ---------------------------------------------------------------------------
template <int K, int NC, bool OUT_F32>
__global__ __launch_bounds__(256) void gemm_bf16(
    const unsigned short* __restrict__ A,    // [ROWS_, K] bf16
    const unsigned short* __restrict__ Wbf,  // [NC, K] bf16
    unsigned short* __restrict__ Hbf,        // out if !OUT_F32 (raw pre-BN)
    float* __restrict__ Hf,                  // out if OUT_F32  (raw pre-BN)
    float* __restrict__ psum,                // [(ROWS/128)*4, NC]
    float* __restrict__ psq)
{
  __shared__ unsigned short As[2][128 * 40]; // +8 pad: 80B stride, b128-aligned
  __shared__ unsigned short Bs[2][64 * 40];
  const int tid  = threadIdx.x;
  const int wave = tid >> 5, lane = tid & 31;
  const int half = lane >> 4, l16 = lane & 15;
  const int mw = wave >> 1, nw = wave & 1;
  const int m0 = blockIdx.x * 128;
  const int n0 = blockIdx.y * 64;

  // Stage one 32-wide K-slice into LDS buffer `bi` (3 async b128 per thread).
  auto stage = [&](int bi, int kk) {
#pragma unroll
    for (int i = 0; i < 2; i++) {            // A tile: 128 rows x 64B
      int e = i * 256 + tid;
      int r = e >> 2, seg = e & 3;
      async_copy_b128(A + (size_t)(m0 + r) * K + kk + seg * 8,
                      &As[bi][r * 40 + seg * 8]);
    }
    {                                        // B tile: 64 rows x 64B
      int r = tid >> 2, seg = tid & 3;
      async_copy_b128(Wbf + (size_t)(n0 + r) * K + kk + seg * 8,
                      &Bs[bi][r * 40 + seg * 8]);
    }
  };

  v8f c00 = {}, c01 = {}, c10 = {}, c11 = {};
  constexpr int T = K / 32;
  stage(0, 0);                               // prologue
  for (int t = 0; t < T; t++) {
    const int buf = t & 1;
    if (t + 1 < T) {
      stage(buf ^ 1, (t + 1) * 32);          // overlap next tile's DMA with math
      wait_async_le3();                      // oldest 3 copies (tile t) done
    } else {
      wait_async0();
    }
    __syncthreads();                         // all waves' tile-t data visible

    union { v16bf v; unsigned int u[8]; } a0, a1, b0, b1;
    const int ar0 = (mw * 32 + l16) * 40;
    const int ar1 = (mw * 32 + 16 + l16) * 40;
#pragma unroll
    for (int j = 0; j < 8; j++) {            // A 16-bit layout: K pairs per VGPR
      int kb = ((j & 3) << 1) + ((j >> 2) << 4) + (half << 3);
      a0.u[j] = *(const unsigned int*)&As[buf][ar0 + kb];
      a1.u[j] = *(const unsigned int*)&As[buf][ar1 + kb];
    }
    const int br0 = (nw * 32 + l16) * 40;
    const int br1 = (nw * 32 + 16 + l16) * 40;
#pragma unroll
    for (int j = 0; j < 8; j++) {            // B 16-bit layout: K pairs per VGPR (same as A)
      int kb = ((j & 3) << 1) + ((j >> 2) << 4) + (half << 3);
      b0.u[j] = *(const unsigned int*)&Bs[buf][br0 + kb];
      b1.u[j] = *(const unsigned int*)&Bs[buf][br1 + kb];
    }
    c00 = __builtin_amdgcn_wmma_f32_16x16x32_bf16(false, a0.v, false, b0.v, (short)0, c00, false, false);
    c01 = __builtin_amdgcn_wmma_f32_16x16x32_bf16(false, a0.v, false, b1.v, (short)0, c01, false, false);
    c10 = __builtin_amdgcn_wmma_f32_16x16x32_bf16(false, a1.v, false, b0.v, (short)0, c10, false, false);
    c11 = __builtin_amdgcn_wmma_f32_16x16x32_bf16(false, a1.v, false, b1.v, (short)0, c11, false, false);
    __syncthreads();                         // all reads of `buf` done before reuse
  }

  const int gn0  = n0 + nw * 32 + l16;
  const int gn1  = gn0 + 16;
  const int gm00 = m0 + mw * 32 + half * 8;       // mi=0 rows
  const int gm10 = gm00 + 16;                     // mi=1 rows
  float ps0 = 0.f, pq0 = 0.f, ps1 = 0.f, pq1 = 0.f;
#pragma unroll
  for (int r = 0; r < 8; r++) {
    float v00 = c00[r], v01 = c01[r], v10 = c10[r], v11 = c11[r];
    if (OUT_F32) {
      Hf[(size_t)(gm00 + r) * NC + gn0] = v00;
      Hf[(size_t)(gm00 + r) * NC + gn1] = v01;
      Hf[(size_t)(gm10 + r) * NC + gn0] = v10;
      Hf[(size_t)(gm10 + r) * NC + gn1] = v11;
    } else {
      Hbf[(size_t)(gm00 + r) * NC + gn0] = f2bf(v00);
      Hbf[(size_t)(gm00 + r) * NC + gn1] = f2bf(v01);
      Hbf[(size_t)(gm10 + r) * NC + gn0] = f2bf(v10);
      Hbf[(size_t)(gm10 + r) * NC + gn1] = f2bf(v11);
    }
    ps0 += v00 + v10;  pq0 += v00 * v00 + v10 * v10;
    ps1 += v01 + v11;  pq1 += v01 * v01 + v11 * v11;
  }
  ps0 += __shfl_xor(ps0, 16);  pq0 += __shfl_xor(pq0, 16);
  ps1 += __shfl_xor(ps1, 16);  pq1 += __shfl_xor(pq1, 16);
  if (half == 0) {                           // unique slot per (Mblock, mw, channel)
    int prow = blockIdx.x * 4 + mw;
    psum[(size_t)prow * NC + gn0] = ps0;  psq[(size_t)prow * NC + gn0] = pq0;
    psum[(size_t)prow * NC + gn1] = ps1;  psq[(size_t)prow * NC + gn1] = pq1;
  }
}

// ---------------------------------------------------------------------------
// Deterministic per-channel reduction of wave partials (fixed-order tree).
// ---------------------------------------------------------------------------
__global__ __launch_bounds__(256) void reduce_stats(
    const float* __restrict__ psum, const float* __restrict__ psq,
    float* __restrict__ sum, float* __restrict__ sq, int nparts, int C)
{
  __shared__ float s1[256], s2[256];
  const int cch = blockIdx.x;
  float a = 0.f, b = 0.f;
  for (int i = threadIdx.x; i < nparts; i += 256) {
    a += psum[(size_t)i * C + cch];
    b += psq [(size_t)i * C + cch];
  }
  s1[threadIdx.x] = a; s2[threadIdx.x] = b;
  __syncthreads();
  for (int st = 128; st > 0; st >>= 1) {
    if (threadIdx.x < st) {
      s1[threadIdx.x] += s1[threadIdx.x + st];
      s2[threadIdx.x] += s2[threadIdx.x + st];
    }
    __syncthreads();
  }
  if (threadIdx.x == 0) { sum[cch] = s1[0]; sq[cch] = s2[0]; }
}

__global__ void make_scalebias(const float* __restrict__ sum, const float* __restrict__ sq,
                               const float* __restrict__ g, const float* __restrict__ b,
                               float* __restrict__ scale, float* __restrict__ bias,
                               int C, float invCount)
{
  int cch = blockIdx.x * blockDim.x + threadIdx.x;
  if (cch < C) {
    float mean = sum[cch] * invCount;
    float var  = sq[cch] * invCount - mean * mean;
    float s    = g[cch] * rsqrtf(var + 1e-5f);
    scale[cch] = s;
    bias[cch]  = b[cch] - mean * s;
  }
}

// f32 -> bf16 bulk convert (weights)
__global__ void f32_to_bf16(const float* __restrict__ src,
                            unsigned short* __restrict__ dst, int n)
{
  int i = blockIdx.x * blockDim.x + threadIdx.x;
  if (i < n) dst[i] = f2bf(src[i]);
}

// BN1 + ReLU applied in place to bf16 h1 (2 elements per uint)
__global__ void bn_relu_bf16_inplace(unsigned int* __restrict__ h,
                                     const float* __restrict__ scale,
                                     const float* __restrict__ bias, int totalPairs)
{
  int i = blockIdx.x * blockDim.x + threadIdx.x;
  if (i < totalPairs) {
    unsigned u = h[i];
    int ch0 = (i * 2) & (C1_ - 1);
    float v0 = fmaxf(fmaf(bf2f((unsigned short)(u & 0xFFFFu)), scale[ch0],     bias[ch0]),     0.f);
    float v1 = fmaxf(fmaf(bf2f((unsigned short)(u >> 16)),     scale[ch0 + 1], bias[ch0 + 1]), 0.f);
    h[i] = (unsigned)f2bf(v0) | ((unsigned)f2bf(v1) << 16);
  }
}

// BN2 + ReLU in place on f32 output
__global__ void bn_relu_inplace(float* __restrict__ out,
                                const float* __restrict__ scale,
                                const float* __restrict__ bias, int total)
{
  int i = blockIdx.x * blockDim.x + threadIdx.x;
  if (i < total) {
    int ch = i & (C2_ - 1);
    out[i] = fmaxf(fmaf(out[i], scale[ch], bias[ch]), 0.f);
  }
}

// ---------------------------------------------------------------------------
extern "C" void kernel_launch(void* const* d_in, const int* in_sizes, int n_in,
                              void* d_out, int out_size, void* d_ws, size_t ws_size,
                              hipStream_t stream) {
  const float* xyz1    = (const float*)d_in[0];
  const float* xyz2    = (const float*)d_in[1];
  const float* points1 = (const float*)d_in[2];
  const float* points2 = (const float*)d_in[3];
  const float* W1      = (const float*)d_in[4];
  const float* g1      = (const float*)d_in[5];
  const float* b1      = (const float*)d_in[6];
  const float* W2      = (const float*)d_in[7];
  const float* g2      = (const float*)d_in[8];
  const float* b2      = (const float*)d_in[9];
  float* out = (float*)d_out;

  char* ws = (char*)d_ws;
  // Workspace layout (all 256B-aligned), total ~82 MB:
  unsigned short* np_bf = (unsigned short*)(ws);             // 65536*320*2 = 41,943,040
  unsigned short* h1    = (unsigned short*)(ws + 41943040);  // 65536*256*2 = 33,554,432
  unsigned short* w1bf  = (unsigned short*)(ws + 75497472);  // 256*320*2   =    163,840
  unsigned short* w2bf  = (unsigned short*)(ws + 75661312);  // 128*256*2   =     65,536
  float* psum1 = (float*)(ws + 75726848);                    // 2048*256*4  =  2,097,152
  float* psq1  = (float*)(ws + 77824000);                    // 2,097,152
  float* psum2 = (float*)(ws + 79921152);                    // 2048*128*4  =  1,048,576
  float* psq2  = (float*)(ws + 80969728);                    // 1,048,576
  float* sc    = (float*)(ws + 82018304);                    // scalars
  float* sum1 = sc,        *sq1 = sc + 256,  *scale1 = sc + 512,  *bias1 = sc + 768;
  float* sum2 = sc + 1024, *sq2 = sc + 1152, *scale2 = sc + 1280, *bias2 = sc + 1408;

  const float invCount = 1.0f / (float)ROWS_;
  const int nparts = (ROWS_ / 128) * 4;      // 2048 partial rows per layer

  interp_concat<<<dim3(N_ / 256, B_), 256, 0, stream>>>(
      xyz1, xyz2, points1, points2, np_bf);

  f32_to_bf16<<<(C1_ * CIN_ + 255) / 256, 256, 0, stream>>>(W1, w1bf, C1_ * CIN_);
  f32_to_bf16<<<(C2_ * C1_  + 255) / 256, 256, 0, stream>>>(W2, w2bf, C2_ * C1_);

  gemm_bf16<CIN_, C1_, false><<<dim3(ROWS_ / 128, C1_ / 64), 256, 0, stream>>>(
      np_bf, w1bf, h1, nullptr, psum1, psq1);
  reduce_stats<<<C1_, 256, 0, stream>>>(psum1, psq1, sum1, sq1, nparts, C1_);
  make_scalebias<<<1, 256, 0, stream>>>(sum1, sq1, g1, b1, scale1, bias1, C1_, invCount);

  const int pairs1 = ROWS_ * C1_ / 2;        // 8,388,608
  bn_relu_bf16_inplace<<<(pairs1 + 255) / 256, 256, 0, stream>>>(
      (unsigned int*)h1, scale1, bias1, pairs1);

  gemm_bf16<C1_, C2_, true><<<dim3(ROWS_ / 128, C2_ / 64), 256, 0, stream>>>(
      h1, w2bf, nullptr, out, psum2, psq2);
  reduce_stats<<<C2_, 256, 0, stream>>>(psum2, psq2, sum2, sq2, nparts, C2_);
  make_scalebias<<<1, 128, 0, stream>>>(sum2, sq2, g2, b2, scale2, bias2, C2_, invCount);

  const int total = ROWS_ * C2_;             // 8,388,608
  bn_relu_inplace<<<(total + 255) / 256, 256, 0, stream>>>(out, scale2, bias2, total);
}
